// ResidualAttentionBlock_28467043238750
// MI455X (gfx1250) — matrix operand order repaired and checked by reading, not compiled, not run on hardware
//
#include <hip/hip_runtime.h>
#include <hip/hip_bf16.h>
#include <math.h>

typedef __attribute__((ext_vector_type(16))) __bf16 v16bf;
typedef __attribute__((ext_vector_type(8)))  __bf16 v8bf;
typedef __attribute__((ext_vector_type(8)))  float  v8f;

#define DM 768
#define NHEAD 12
#define HD 64

__device__ __forceinline__ unsigned short f2bf(float f) {
  unsigned u = __builtin_bit_cast(unsigned, f);
  u += 0x7fffu + ((u >> 16) & 1u);
  return (unsigned short)(u >> 16);
}
__device__ __forceinline__ float bf2f(unsigned short h) {
  unsigned u = (unsigned)h << 16;
  return __builtin_bit_cast(float, u);
}
__device__ __forceinline__ v8f vz() {
  v8f z;
  for (int i = 0; i < 8; i++) z[i] = 0.f;
  return z;
}

// Workgroup-relative LDS byte offset: generic pointers to LDS carry the
// LDS offset in their low 32 bits (flat LDS aperture).
__device__ __forceinline__ unsigned lds_off(const void* p) {
  return (unsigned)(unsigned long long)p;
}
// CDNA5 async DMA global->LDS, 16B per lane, tracked by ASYNCcnt.
__device__ __forceinline__ void async_b128(unsigned ldsoff, const void* g) {
  asm volatile("global_load_async_to_lds_b128 %0, %1, off"
               :: "v"(ldsoff), "v"(g) : "memory");
}
#define ASYNC_WAIT(N) asm volatile("s_wait_asynccnt " #N ::: "memory")

// ---------------------------------------------------------------------------
// Generic bf16 GEMM:  C[M,N] = act(A[M,K] * W[N,K]^T + bias)
// ACT: 0=none, 1=gelu(erf), 2=x*sigmoid(1.702x)
// OMODE: 0=write f32, 1=write bf16, 2=accumulate into f32
// Block 256 thr = 8 waves; block tile 256(M) x 64(N); per-wave 32x64 tile
// (8 WMMA per K-step, A-operand reuse across 4 N-chunks). A/B tiles
// double-buffered in LDS via global_load_async_to_lds_b128 (5 per thread).
// ---------------------------------------------------------------------------
template<int ACT, int OMODE>
__global__ void __launch_bounds__(256) gemm_bf16(
    const unsigned short* __restrict__ A,
    const unsigned short* __restrict__ W,
    const float* __restrict__ bias,
    void* __restrict__ Cout,
    int M, int N, int K)
{
  __shared__ unsigned short ta[2][256 * 32];   // 16KB per buffer
  __shared__ unsigned short tb[2][64 * 32];    // 4KB per buffer

  int tid  = threadIdx.x;
  int wave = tid >> 5, lane = tid & 31;
  int l16  = lane & 15, lh = lane >> 4;
  long ablk  = (long)blockIdx.x * 256;          // block A row base
  long bblk  = (long)blockIdx.y * 64;           // block W row base
  long mbase = ablk + wave * 32;

  v8f acc[2][4];
#pragma unroll
  for (int mi = 0; mi < 2; mi++)
#pragma unroll
    for (int nc = 0; nc < 4; nc++) acc[mi][nc] = vz();

  auto issue_tile = [&](int kt, int p) {
    int kk = kt << 5;
    // A: each thread copies one full 32-elem row (64B, 4 contiguous b128)
    const unsigned short* ar = A + (size_t)(ablk + tid) * K + kk;
    unsigned al = lds_off(&ta[p][tid * 32]);
    async_b128(al,      ar);
    async_b128(al + 16, ar + 8);
    async_b128(al + 32, ar + 16);
    async_b128(al + 48, ar + 24);
    // B: 256 x 16B chunks, one per thread
    int row = tid >> 2, col8 = tid & 3;
    async_b128(lds_off(&tb[p][row * 32 + col8 * 8]),
               W + (size_t)(bblk + row) * K + kk + col8 * 8);
  };

  int nk = K >> 5;
  issue_tile(0, 0);
  for (int kt = 0; kt < nk; kt++) {
    int p = kt & 1;
    if (kt + 1 < nk) {
      issue_tile(kt + 1, p ^ 1);
      ASYNC_WAIT(5);       // current tile complete, next tile in flight
    } else {
      ASYNC_WAIT(0);
    }
    __syncthreads();

    const unsigned short* ap = &ta[p][0] + (wave * 32) * 32;
    const unsigned short* bp = &tb[p][0];
    v16bf a0, a1;
    {
      v8bf lo0 = *(const v8bf*)(ap + l16 * 32 + lh * 16);
      v8bf hi0 = *(const v8bf*)(ap + l16 * 32 + lh * 16 + 8);
      v8bf lo1 = *(const v8bf*)(ap + (16 + l16) * 32 + lh * 16);
      v8bf hi1 = *(const v8bf*)(ap + (16 + l16) * 32 + lh * 16 + 8);
#pragma unroll
      for (int i = 0; i < 8; i++) {
        a0[i] = lo0[i]; a0[i + 8] = hi0[i];
        a1[i] = lo1[i]; a1[i + 8] = hi1[i];
      }
    }
#pragma unroll
    for (int nc = 0; nc < 4; nc++) {
      v16bf b = *(const v16bf*)(bp + (nc * 16 + l16) * 32 + lh * 16);
      acc[0][nc] = __builtin_amdgcn_wmma_f32_16x16x32_bf16(false, a0, false, b, (short)0, acc[0][nc], false, false);
      acc[1][nc] = __builtin_amdgcn_wmma_f32_16x16x32_bf16(false, a1, false, b, (short)0, acc[1][nc], false, false);
    }
    __syncthreads();       // all waves done reading buf p before it is reloaded
  }

#pragma unroll
  for (int mi = 0; mi < 2; mi++)
#pragma unroll
    for (int nc = 0; nc < 4; nc++) {
      v8f a = acc[mi][nc];
      long col = bblk + nc * 16 + l16;
      float bv = bias ? bias[col] : 0.f;
#pragma unroll
      for (int j = 0; j < 8; j++) {
        long row = mbase + mi * 16 + j + lh * 8;
        float v = a[j] + bv;
        if (ACT == 1)      v = 0.5f * v * (1.0f + erff(v * 0.70710678f));
        else if (ACT == 2) v = v / (1.0f + __expf(-1.702f * v));
        size_t idx = (size_t)row * N + col;
        if (OMODE == 0)      ((float*)Cout)[idx] = v;
        else if (OMODE == 1) ((unsigned short*)Cout)[idx] = f2bf(v);
        else                 ((float*)Cout)[idx] += v;
      }
    }
}

// ---------------------------------------------------------------------------
// Flash attention over one (sequence n, head h) per workgroup.
// qkv rows: [n*L + l][2304], q at h*64, k at 768+h*64, v at 1536+h*64.
// out bf16: [n*L + l][768] at h*64.
// K staged by async DMA (layout matches global); V vector-loaded + transposed.
// ---------------------------------------------------------------------------
__global__ void __launch_bounds__(256) attn_flash(
    const unsigned short* __restrict__ qkv,
    unsigned short* __restrict__ out,
    const float* __restrict__ mask,
    int L, int Lpad, int maskmod)
{
  extern __shared__ unsigned short smem[];
  unsigned short* kls = smem;                  // [Lpad][64] row-major
  unsigned short* vls = smem + Lpad * HD;      // [64][Lpad] (transposed)
  unsigned short* pls = smem + 2 * Lpad * HD;  // [8 waves][16][32]

  int bid  = blockIdx.x;
  int head = bid % NHEAD;
  int n    = bid / NHEAD;
  const int ldq = 3 * DM;
  const unsigned short* qbase = qkv + (size_t)n * L * ldq + head * HD;
  const unsigned short* kbase = qbase + DM;
  const unsigned short* vbase = qbase + 2 * DM;
  const float* maskb = mask ? (mask + (size_t)(n % maskmod) * L * L) : (const float*)0;

  int tid = threadIdx.x;
  unsigned klsoff = lds_off(kls);
  for (int ch = tid; ch < Lpad * 8; ch += 256) {   // K: async b128 copies
    int l = ch >> 3, c8 = ch & 7;
    int lc = l < L ? l : L - 1;                    // pad rows masked later
    async_b128(klsoff + (unsigned)(l * HD + c8 * 8) * 2,
               kbase + (size_t)lc * ldq + c8 * 8);
  }
  for (int ch = tid; ch < Lpad * 8; ch += 256) {   // V: vector load + transpose
    int l = ch >> 3, c8 = ch & 7;
    int lc = l < L ? l : L - 1;
    v8bf vv = *(const v8bf*)(vbase + (size_t)lc * ldq + c8 * 8);
    for (int i = 0; i < 8; i++)
      vls[(c8 * 8 + i) * Lpad + l] = __builtin_bit_cast(unsigned short, vv[i]);
  }
  ASYNC_WAIT(0);
  __syncthreads();

  int wave = tid >> 5, lane = tid & 31;
  int l16 = lane & 15, lh = lane >> 4;
  int ntiles  = (L + 15) >> 4;
  int nchunks = Lpad >> 5;

  for (int rt = wave; rt < ntiles; rt += 8) {
    int r0 = rt << 4;
    int qrow = r0 + l16; if (qrow >= L) qrow = L - 1;
    const unsigned short* qp = qbase + (size_t)qrow * ldq + lh * 16;
    v16bf Aq0, Aq1;
    {
      v8bf x0 = *(const v8bf*)(qp);
      v8bf x1 = *(const v8bf*)(qp + 8);
      v8bf x2 = *(const v8bf*)(qp + 32);
      v8bf x3 = *(const v8bf*)(qp + 40);
      for (int i = 0; i < 8; i++) {
        Aq0[i] = x0[i]; Aq0[i + 8] = x1[i];
        Aq1[i] = x2[i]; Aq1[i + 8] = x3[i];
      }
    }
    v8f O0 = vz(), O1 = vz(), O2 = vz(), O3 = vz();
    float mr[8], sr[8];
    for (int j = 0; j < 8; j++) { mr[j] = -3.0e38f; sr[j] = 0.f; }

    unsigned short* pw = pls + wave * 512;

    for (int cc = 0; cc < nchunks; cc++) {
      v8f S0 = vz(), S1 = vz();
      {
        int cb = (cc << 5) + l16;
        v16bf B00 = *(const v16bf*)(&kls[cb * HD + lh * 16]);
        v16bf B01 = *(const v16bf*)(&kls[cb * HD + 32 + lh * 16]);
        v16bf B10 = *(const v16bf*)(&kls[(cb + 16) * HD + lh * 16]);
        v16bf B11 = *(const v16bf*)(&kls[(cb + 16) * HD + 32 + lh * 16]);
        S0 = __builtin_amdgcn_wmma_f32_16x16x32_bf16(false, Aq0, false, B00, (short)0, S0, false, false);
        S0 = __builtin_amdgcn_wmma_f32_16x16x32_bf16(false, Aq1, false, B01, (short)0, S0, false, false);
        S1 = __builtin_amdgcn_wmma_f32_16x16x32_bf16(false, Aq0, false, B10, (short)0, S1, false, false);
        S1 = __builtin_amdgcn_wmma_f32_16x16x32_bf16(false, Aq1, false, B11, (short)0, S1, false, false);
      }
      float p0a[8], p1a[8], efj[8];
      int c0 = (cc << 5) + l16, c1 = c0 + 16;
      for (int j = 0; j < 8; j++) {
        int row  = r0 + j + lh * 8;
        int rowc = row < L ? row : (L - 1);
        float s0 = S0[j] * 0.125f;
        float s1 = S1[j] * 0.125f;
        if (maskb) {
          if (c0 < L) s0 += maskb[(size_t)rowc * L + c0];
          if (c1 < L) s1 += maskb[(size_t)rowc * L + c1];
        }
        if (c0 >= L) s0 = -3.0e38f;
        if (c1 >= L) s1 = -3.0e38f;
        float cm = fmaxf(s0, s1);
        for (int dd = 1; dd < 16; dd <<= 1) cm = fmaxf(cm, __shfl_xor(cm, dd, 32));
        float nm = fmaxf(mr[j], cm);
        float ef = __expf(mr[j] - nm);
        float p0 = __expf(s0 - nm);
        float p1 = __expf(s1 - nm);
        float rs = p0 + p1;
        for (int dd = 1; dd < 16; dd <<= 1) rs += __shfl_xor(rs, dd, 32);
        sr[j] = sr[j] * ef + rs;
        mr[j] = nm;
        p0a[j] = p0; p1a[j] = p1; efj[j] = ef;
      }
      for (int j = 0; j < 8; j++) {
        O0[j] *= efj[j]; O1[j] *= efj[j]; O2[j] *= efj[j]; O3[j] *= efj[j];
      }
      // stage P (C-layout -> A-layout) through LDS, intra-wave only
      for (int j = 0; j < 8; j++) {
        int rl = j + lh * 8;
        pw[rl * 32 + l16]      = f2bf(p0a[j]);
        pw[rl * 32 + 16 + l16] = f2bf(p1a[j]);
      }
      v16bf Pa;
      {
        v8bf plo = *(const v8bf*)(pw + l16 * 32 + lh * 16);
        v8bf phi = *(const v8bf*)(pw + l16 * 32 + lh * 16 + 8);
        for (int i = 0; i < 8; i++) { Pa[i] = plo[i]; Pa[i + 8] = phi[i]; }
      }
      {
        int kb = (cc << 5) + lh * 16;
        v16bf Bv0 = *(const v16bf*)(&vls[(0 * 16 + l16) * Lpad + kb]);
        v16bf Bv1 = *(const v16bf*)(&vls[(1 * 16 + l16) * Lpad + kb]);
        v16bf Bv2 = *(const v16bf*)(&vls[(2 * 16 + l16) * Lpad + kb]);
        v16bf Bv3 = *(const v16bf*)(&vls[(3 * 16 + l16) * Lpad + kb]);
        O0 = __builtin_amdgcn_wmma_f32_16x16x32_bf16(false, Pa, false, Bv0, (short)0, O0, false, false);
        O1 = __builtin_amdgcn_wmma_f32_16x16x32_bf16(false, Pa, false, Bv1, (short)0, O1, false, false);
        O2 = __builtin_amdgcn_wmma_f32_16x16x32_bf16(false, Pa, false, Bv2, (short)0, O2, false, false);
        O3 = __builtin_amdgcn_wmma_f32_16x16x32_bf16(false, Pa, false, Bv3, (short)0, O3, false, false);
      }
    }
    for (int j = 0; j < 8; j++) {
      int row = r0 + j + lh * 8;
      if (row < L) {
        float inv = 1.0f / sr[j];
        size_t ob = ((size_t)(n * L + row)) * DM + head * HD;
        out[ob + 0 * 16 + l16] = f2bf(O0[j] * inv);
        out[ob + 1 * 16 + l16] = f2bf(O1[j] * inv);
        out[ob + 2 * 16 + l16] = f2bf(O2[j] * inv);
        out[ob + 3 * 16 + l16] = f2bf(O3[j] * inv);
      }
    }
  }
}

// ---------------------------------------------------------------------------
// LayerNorm over 768, one token per 256-thread block, output bf16.
// ---------------------------------------------------------------------------
__device__ __forceinline__ void ln_token(const float* __restrict__ src,
    const float* __restrict__ g, const float* __restrict__ b,
    unsigned short* __restrict__ dst)
{
  __shared__ float red[16];
  int tid = threadIdx.x;
  float v0 = src[tid], v1 = src[tid + 256], v2 = src[tid + 512];
  float s = v0 + v1 + v2;
  float q = v0 * v0 + v1 * v1 + v2 * v2;
  for (int dd = 1; dd < 32; dd <<= 1) {
    s += __shfl_xor(s, dd, 32);
    q += __shfl_xor(q, dd, 32);
  }
  if ((tid & 31) == 0) { red[tid >> 5] = s; red[8 + (tid >> 5)] = q; }
  __syncthreads();
  float ts = 0.f, tq = 0.f;
  for (int w = 0; w < 8; w++) { ts += red[w]; tq += red[8 + w]; }
  float mean = ts * (1.0f / 768.0f);
  float var  = tq * (1.0f / 768.0f) - mean * mean;
  float inv  = rsqrtf(var + 1e-5f);
  dst[tid]       = f2bf((v0 - mean) * inv * g[tid]       + b[tid]);
  dst[tid + 256] = f2bf((v1 - mean) * inv * g[tid + 256] + b[tid + 256]);
  dst[tid + 512] = f2bf((v2 - mean) * inv * g[tid + 512] + b[tid + 512]);
}

// shifted-window gather + LN: token = n*128+l, n = b*49+wh*7+ww, l = t*4+ih*2+iw
__global__ void __launch_bounds__(256) ln_window(const float* __restrict__ x,
    const float* __restrict__ g, const float* __restrict__ b,
    unsigned short* __restrict__ out)
{
  int token = blockIdx.x;
  int n = token >> 7, l = token & 127;
  int bb = n / 49, wrem = n % 49;
  int wh = wrem / 7, ww = wrem % 7;
  int t = l >> 2, ih = (l >> 1) & 1, iw = l & 1;
  int hs  = (2 * wh + ih + 1) % 14;   // forward shift (-1,-1)
  int wsp = (2 * ww + iw + 1) % 14;
  const float* src = x + ((size_t)(1 + hs * 14 + wsp) * 256 + (bb * 32 + t)) * DM;
  ln_token(src, g, b, out + (size_t)token * DM);
}

__global__ void __launch_bounds__(256) ln_plain(const float* __restrict__ src,
    const float* __restrict__ g, const float* __restrict__ b,
    unsigned short* __restrict__ out)
{
  size_t r = blockIdx.x;
  ln_token(src + r * DM, g, b, out + r * DM);
}

// x2 = [x[0], cls_attn, x[1:]] gathered into (l,n) order: r = n*198 + l
__global__ void __launch_bounds__(256) ln_space(const float* __restrict__ xcur,
    const float* __restrict__ cls,
    const float* __restrict__ g, const float* __restrict__ b,
    unsigned short* __restrict__ out)
{
  int r = blockIdx.x;
  int n = r / 198, l = r % 198;
  const float* src;
  if (l == 1) src = cls + (size_t)n * DM;
  else {
    int s = (l == 0) ? 0 : (l - 1);
    src = xcur + ((size_t)s * 256 + n) * DM;
  }
  ln_token(src, g, b, out + (size_t)r * DM);
}

// window_reverse + roll(+1,+1) scatter into xt rows 1..196
__global__ void __launch_bounds__(256) scatter_win(
    const unsigned short* __restrict__ proj, unsigned short* __restrict__ xt)
{
  int token = blockIdx.x;            // p*256 + j, p in 0..195
  int p = token >> 8, j = token & 255;
  int hs = p / 14, wsp = p % 14;
  int bb = j >> 5, t = j & 31;
  int hp = (hs + 13) % 14, wp = (wsp + 13) % 14;
  int wh = hp >> 1, ih = hp & 1, ww = wp >> 1, iw = wp & 1;
  int n = bb * 49 + wh * 7 + ww, l = t * 4 + ih * 2 + iw;
  const unsigned short* s = proj + ((size_t)(n * 128 + l)) * DM;
  unsigned short* d = xt + ((size_t)((p + 1) * 256 + j)) * DM;
  for (int c = threadIdx.x; c < DM; c += 256) d[c] = s[c];
}

__global__ void __launch_bounds__(256) cls_to_xt(
    const float* __restrict__ cls, unsigned short* __restrict__ xt)
{
  size_t r = blockIdx.x;
  for (int c = threadIdx.x; c < DM; c += 256)
    xt[r * DM + c] = f2bf(cls[r * DM + c]);
}

// x += proj + adapter_out for surviving rows of the 198-token sequence
__global__ void __launch_bounds__(256) scatter_space(
    const unsigned short* __restrict__ projb, const float* __restrict__ adap,
    float* __restrict__ xcur)
{
  int token = blockIdx.x;            // s*256 + j, s in 0..196
  int sidx = token >> 8, j = token & 255;
  int s2 = (sidx == 0) ? 0 : (sidx + 1);
  size_t rs = (size_t)(j * 198 + s2) * DM;
  float* d = xcur + (size_t)token * DM;
  for (int c = threadIdx.x; c < DM; c += 256)
    d[c] += bf2f(projb[rs + c]) + adap[rs + c];
}

__global__ void __launch_bounds__(256) f2bf_convert(
    const float* __restrict__ s, unsigned short* __restrict__ d, int n)
{
  int i = blockIdx.x * 256 + threadIdx.x;
  if (i < n) d[i] = f2bf(s[i]);
}

// ---------------------------------------------------------------------------
extern "C" void kernel_launch(void* const* d_in, const int* in_sizes, int n_in,
                              void* d_out, int out_size, void* d_ws, size_t ws_size,
                              hipStream_t stream)
{
  (void)in_sizes; (void)n_in; (void)out_size; (void)ws_size;
  const float* x     = (const float*)d_in[0];
  const float* mask  = (const float*)d_in[1];
  const float* in_w  = (const float*)d_in[2];
  const float* in_b  = (const float*)d_in[3];
  const float* out_w = (const float*)d_in[4];
  const float* out_b = (const float*)d_in[5];
  const float* ln1_g = (const float*)d_in[6];
  const float* ln1_b = (const float*)d_in[7];
  const float* ln2_g = (const float*)d_in[8];
  const float* ln2_b = (const float*)d_in[9];
  const float* fc_w  = (const float*)d_in[10];
  const float* fc_b  = (const float*)d_in[11];
  const float* pr_w  = (const float*)d_in[12];
  const float* pr_b  = (const float*)d_in[13];
  const float* sa1_w = (const float*)d_in[14];
  const float* sa1_b = (const float*)d_in[15];
  const float* sa2_w = (const float*)d_in[16];
  const float* sa2_b = (const float*)d_in[17];
  const float* ta1_w = (const float*)d_in[18];
  const float* ta1_b = (const float*)d_in[19];
  const float* ta2_w = (const float*)d_in[20];
  const float* ta2_b = (const float*)d_in[21];
  const float* ma1_w = (const float*)d_in[22];
  const float* ma1_b = (const float*)d_in[23];
  const float* ma2_w = (const float*)d_in[24];
  const float* ma2_b = (const float*)d_in[25];

  const int M_WIN = 392 * 128;   // 50176 window tokens
  const int M_CLS = 256;         // cls tokens
  const int M_SPC = 256 * 198;   // 50688 padded-seq tokens
  const int M_SEQ = 197 * 256;   // 50432 output tokens

  size_t off = 0;
  auto alloc = [&](size_t bytes) -> void* {
    void* p = (char*)d_ws + off;
    off += (bytes + 255) & ~(size_t)255;
    return p;
  };
  unsigned short* wb_in  = (unsigned short*)alloc((size_t)2304 * 768 * 2);
  unsigned short* wb_out = (unsigned short*)alloc((size_t)768 * 768 * 2);
  unsigned short* wb_fc  = (unsigned short*)alloc((size_t)3072 * 768 * 2);
  unsigned short* wb_pr  = (unsigned short*)alloc((size_t)768 * 3072 * 2);
  unsigned short* wb_sa1 = (unsigned short*)alloc((size_t)192 * 768 * 2);
  unsigned short* wb_sa2 = (unsigned short*)alloc((size_t)768 * 192 * 2);
  unsigned short* wb_ta1 = (unsigned short*)alloc((size_t)192 * 768 * 2);
  unsigned short* wb_ta2 = (unsigned short*)alloc((size_t)768 * 192 * 2);
  unsigned short* wb_ma1 = (unsigned short*)alloc((size_t)192 * 768 * 2);
  unsigned short* wb_ma2 = (unsigned short*)alloc((size_t)768 * 192 * 2);
  unsigned short* Abuf = (unsigned short*)alloc((size_t)M_SPC * 768 * 2);
  size_t qsz = (size_t)M_SPC * 2304 * 2, hsz = (size_t)M_SEQ * 3072 * 2;
  unsigned short* Qbuf = (unsigned short*)alloc(qsz > hsz ? qsz : hsz);
  unsigned short* Hbuf = Qbuf;   // MLP hidden overlays dead QKV buffer
  unsigned short* Cbuf = (unsigned short*)alloc((size_t)M_SPC * 768 * 2);
  unsigned short* Dbuf = (unsigned short*)alloc((size_t)M_SPC * 768 * 2);
  unsigned short* Ebuf = (unsigned short*)alloc((size_t)M_SPC * 192 * 2);
  float*          Fbuf = (float*)alloc((size_t)M_SPC * 768 * 4);
  unsigned short* XT   = (unsigned short*)alloc((size_t)M_SEQ * 768 * 2);
  float*          CLS  = (float*)alloc((size_t)256 * 768 * 4);

  float* xcur = (float*)d_out;
  hipMemcpyAsync(xcur, x, (size_t)197 * 256 * 768 * 4, hipMemcpyDeviceToDevice, stream);

  auto cvt = [&](const float* s, unsigned short* d, int nels) {
    f2bf_convert<<<dim3((nels + 255) / 256), dim3(256), 0, stream>>>(s, d, nels);
  };
  cvt(in_w, wb_in, 2304 * 768);  cvt(out_w, wb_out, 768 * 768);
  cvt(fc_w, wb_fc, 3072 * 768);  cvt(pr_w, wb_pr, 768 * 3072);
  cvt(sa1_w, wb_sa1, 192 * 768); cvt(sa2_w, wb_sa2, 768 * 192);
  cvt(ta1_w, wb_ta1, 192 * 768); cvt(ta2_w, wb_ta2, 768 * 192);
  cvt(ma1_w, wb_ma1, 192 * 768); cvt(ma2_w, wb_ma2, 768 * 192);

  // Phase A: shifted-window attention
  ln_window<<<dim3(M_WIN), dim3(256), 0, stream>>>(x, ln1_g, ln1_b, Abuf);
  gemm_bf16<0,1><<<dim3(M_WIN/256, 2304/64), dim3(256), 0, stream>>>(Abuf, wb_in, in_b, Qbuf, M_WIN, 2304, 768);
  attn_flash<<<dim3(392*12), dim3(256), (size_t)128*256 + 8192, stream>>>(Qbuf, Cbuf, mask, 128, 128, 49);
  gemm_bf16<0,1><<<dim3(M_WIN/256, 768/64), dim3(256), 0, stream>>>(Cbuf, wb_out, out_b, Dbuf, M_WIN, 768, 768);
  scatter_win<<<dim3(196*256), dim3(256), 0, stream>>>(Dbuf, XT);

  // Phase B: cls attention
  ln_plain<<<dim3(M_CLS), dim3(256), 0, stream>>>(x, ln1_g, ln1_b, Abuf);
  gemm_bf16<0,1><<<dim3(M_CLS/256, 2304/64), dim3(256), 0, stream>>>(Abuf, wb_in, in_b, Qbuf, M_CLS, 2304, 768);
  attn_flash<<<dim3(8*12), dim3(256), (size_t)32*256 + 8192, stream>>>(Qbuf, Cbuf, (const float*)0, 32, 32, 1);
  gemm_bf16<0,0><<<dim3(M_CLS/256, 768/64), dim3(256), 0, stream>>>(Cbuf, wb_out, out_b, CLS, M_CLS, 768, 768);
  cls_to_xt<<<dim3(256), dim3(256), 0, stream>>>(CLS, XT);

  // Phase C: temporal adapter  x += gelu(xt@ta1^T+b1)@ta2^T+b2
  gemm_bf16<1,1><<<dim3(M_SEQ/256, 192/64), dim3(256), 0, stream>>>(XT, wb_ta1, ta1_b, Ebuf, M_SEQ, 192, 768);
  gemm_bf16<0,2><<<dim3(M_SEQ/256, 768/64), dim3(256), 0, stream>>>(Ebuf, wb_ta2, ta2_b, xcur, M_SEQ, 768, 192);

  // Phase D: spatial attention (L=198) + skip-adapter
  ln_space<<<dim3(M_SPC), dim3(256), 0, stream>>>(xcur, CLS, ln1_g, ln1_b, Abuf);
  gemm_bf16<0,1><<<dim3(M_SPC/256, 2304/64), dim3(256), 0, stream>>>(Abuf, wb_in, in_b, Qbuf, M_SPC, 2304, 768);
  attn_flash<<<dim3(256*12), dim3(256), (size_t)224*256 + 8192, stream>>>(Qbuf, Cbuf, (const float*)0, 198, 224, 1);
  gemm_bf16<0,1><<<dim3(M_SPC/256, 768/64), dim3(256), 0, stream>>>(Cbuf, wb_out, out_b, Dbuf, M_SPC, 768, 768);
  gemm_bf16<1,1><<<dim3(M_SPC/256, 192/64), dim3(256), 0, stream>>>(Dbuf, wb_sa1, sa1_b, Ebuf, M_SPC, 192, 768);
  gemm_bf16<0,0><<<dim3(M_SPC/256, 768/64), dim3(256), 0, stream>>>(Ebuf, wb_sa2, sa2_b, Fbuf, M_SPC, 768, 192);
  scatter_space<<<dim3(197*256), dim3(256), 0, stream>>>(Dbuf, Fbuf, xcur);

  // Phase E: MLP (silu-1.702 gate) + modality adapter, both from ln2(x)
  ln_plain<<<dim3(M_SEQ), dim3(256), 0, stream>>>(xcur, ln2_g, ln2_b, Abuf);
  gemm_bf16<2,1><<<dim3(M_SEQ/256, 3072/64), dim3(256), 0, stream>>>(Abuf, wb_fc, fc_b, Hbuf, M_SEQ, 3072, 768);
  gemm_bf16<0,2><<<dim3(M_SEQ/256, 768/64), dim3(256), 0, stream>>>(Hbuf, wb_pr, pr_b, xcur, M_SEQ, 768, 3072);
  gemm_bf16<1,1><<<dim3(M_SEQ/256, 192/64), dim3(256), 0, stream>>>(Abuf, wb_ma1, ma1_b, Ebuf, M_SEQ, 192, 768);
  gemm_bf16<0,2><<<dim3(M_SEQ/256, 768/64), dim3(256), 0, stream>>>(Ebuf, wb_ma2, ma2_b, xcur, M_SEQ, 768, 192);
}
